// PlainGCN_15607911154259
// MI455X (gfx1250) — matrix operand order repaired and checked by reading, not compiled, not run on hardware
//
#include <hip/hip_runtime.h>
#include <stdint.h>

#define NN 8192
#define DD 256
#define LALPHA 0.2f

typedef __bf16 bf16x16 __attribute__((ext_vector_type(16)));
typedef float v8f __attribute__((ext_vector_type(8)));
typedef unsigned int uint4v __attribute__((ext_vector_type(4)));
typedef unsigned int uint2v __attribute__((ext_vector_type(2)));

union ABFrag { bf16x16 v; uint4v q[2]; };

__device__ __forceinline__ unsigned short f2bf(float f) {
    union { float f; unsigned u; } x; x.f = f;
    unsigned u = x.u;
    u += 0x7FFFu + ((u >> 16) & 1u);   // round-to-nearest-even
    return (unsigned short)(u >> 16);
}

// ---- convert inputs X (f32) -> Xb (bf16), 4 elems/thread ----
__global__ void k_cvt_x(const float* __restrict__ X, unsigned short* __restrict__ Xb) {
    size_t i = (size_t)(blockIdx.x * 256 + threadIdx.x) * 4;
    float4 v = *(const float4*)(X + i);
    uint2v u;
    u[0] = (unsigned)f2bf(v.x) | ((unsigned)f2bf(v.y) << 16);
    u[1] = (unsigned)f2bf(v.z) | ((unsigned)f2bf(v.w) << 16);
    *(uint2v*)(Xb + i) = u;
}

// ---- W (f32 [D][D]) -> WbT (bf16, transposed: WbT[n][k] = W[k][n]) ----
__global__ void k_cvt_wT(const float* __restrict__ W, unsigned short* __restrict__ WbT) {
    int n = blockIdx.x;
    int k = threadIdx.x;
    WbT[n * DD + k] = f2bf(W[k * DD + n]);
}

// ---- GEMM1: Wh = X @ W ; also emit WhbT (bf16, transposed) ----
// block = 16 rows x 256 cols, 8 waves, 2 WMMA tiles per wave
__global__ void __launch_bounds__(256) k_gemm1(const unsigned short* __restrict__ Xb,
                                               const unsigned short* __restrict__ WbT,
                                               float* __restrict__ Wh,
                                               unsigned short* __restrict__ WhbT) {
    const int tid  = threadIdx.x;
    const int lane = tid & 31;
    const int wid  = tid >> 5;
    const int hi   = lane >> 4;     // 0/1: which K-half this lane holds
    const int l15  = lane & 15;
    const int row0 = blockIdx.x * 16;
    const int c0   = wid * 32;

    v8f acc0 = {}; v8f acc1 = {};
    const int arow  = row0 + l15;
    const int bcol0 = c0 + l15;
    const int bcol1 = c0 + 16 + l15;

    for (int k0 = 0; k0 < DD; k0 += 32) {
        ABFrag a, b0, b1;
        // A 16x32 bf16: lanes 0-15 hold K 0..7 & 16..23; lanes 16-31 hold K 8..15 & 24..31
        const unsigned short* ap = Xb + arow * DD + k0 + hi * 8;
        a.q[0] = *(const uint4v*)(ap);
        a.q[1] = *(const uint4v*)(ap + 16);
        // B 32x16 bf16: lane = column, lanes 0-15 K 0..7 & 16..23, lanes 16-31 K 8..15 & 24..31
        const unsigned short* bp0 = WbT + bcol0 * DD + k0 + hi * 8;
        b0.q[0] = *(const uint4v*)(bp0);
        b0.q[1] = *(const uint4v*)(bp0 + 16);
        const unsigned short* bp1 = WbT + bcol1 * DD + k0 + hi * 8;
        b1.q[0] = *(const uint4v*)(bp1);
        b1.q[1] = *(const uint4v*)(bp1 + 16);
        acc0 = __builtin_amdgcn_wmma_f32_16x16x32_bf16(false, a.v, false, b0.v,
                                                       (short)0, acc0, false, false);
        acc1 = __builtin_amdgcn_wmma_f32_16x16x32_bf16(false, a.v, false, b1.v,
                                                       (short)0, acc1, false, false);
    }

    // C layout: VGPR v -> (M = v + hi*8, N = l15)
#pragma unroll
    for (int v = 0; v < 8; ++v) {
        int m = v + hi * 8;
        Wh[(size_t)(row0 + m) * DD + bcol0] = acc0[v];
        Wh[(size_t)(row0 + m) * DD + bcol1] = acc1[v];
    }
    // each lane holds 8 consecutive rows of one column -> one packed 16B bf16 store
    uint4v p0, p1;
#pragma unroll
    for (int i = 0; i < 4; ++i) {
        p0[i] = (unsigned)f2bf(acc0[2 * i]) | ((unsigned)f2bf(acc0[2 * i + 1]) << 16);
        p1[i] = (unsigned)f2bf(acc1[2 * i]) | ((unsigned)f2bf(acc1[2 * i + 1]) << 16);
    }
    *(uint4v*)(WhbT + (size_t)bcol0 * NN + row0 + hi * 8) = p0;
    *(uint4v*)(WhbT + (size_t)bcol1 * NN + row0 + hi * 8) = p1;
}

// ---- f1 = Wh@a1, f2 = Wh@a2 (one wave per row) ----
__global__ void k_f12(const float* __restrict__ Wh, const float* __restrict__ a1,
                      const float* __restrict__ a2, float* __restrict__ f1,
                      float* __restrict__ f2) {
    int lane = threadIdx.x & 31;
    int wid  = threadIdx.x >> 5;
    int row  = blockIdx.x * 8 + wid;
    const float* wr = Wh + (size_t)row * DD;
    float s1 = 0.f, s2 = 0.f;
    for (int i = lane; i < DD; i += 32) {
        float w = wr[i];
        s1 += w * a1[i];
        s2 += w * a2[i];
    }
#pragma unroll
    for (int off = 16; off; off >>= 1) {
        s1 += __shfl_xor(s1, off, 32);
        s2 += __shfl_xor(s2, off, 32);
    }
    if (lane == 0) { f1[row] = s1; f2[row] = s2; }
}

// helpers for the fused attention kernel -------------------------------------
__device__ __forceinline__ void consume_tile(const unsigned short* abase,
                                             const unsigned short* bbase0,
                                             const unsigned short* bbase1,
                                             int j0, v8f& acc0, v8f& acc1) {
    ABFrag a, b0, b1;
    a.q[0] = *(const uint4v*)(abase);        // K 0..7   (or 8..15)
    a.q[1] = *(const uint4v*)(abase + 16);   // K 16..23 (or 24..31)
    b0.q[0] = *(const uint4v*)(bbase0 + j0);
    b0.q[1] = *(const uint4v*)(bbase0 + j0 + 16);
    b1.q[0] = *(const uint4v*)(bbase1 + j0);
    b1.q[1] = *(const uint4v*)(bbase1 + j0 + 16);
    acc0 = __builtin_amdgcn_wmma_f32_16x16x32_bf16(false, a.v, false, b0.v,
                                                   (short)0, acc0, false, false);
    acc1 = __builtin_amdgcn_wmma_f32_16x16x32_bf16(false, a.v, false, b1.v,
                                                   (short)0, acc1, false, false);
}

__device__ __forceinline__ void produce_tile(unsigned int* dst, float2 ad,
                                             float g0, float g1, float f1v,
                                             float& lsum) {
    float s0 = f1v + g0; s0 = (s0 > 0.f) ? s0 : LALPHA * s0;
    float s1 = f1v + g1; s1 = (s1 > 0.f) ? s1 : LALPHA * s1;
    float p0 = (ad.x > 0.f) ? __expf(s0) : 0.f;
    float p1 = (ad.y > 0.f) ? __expf(s1) : 0.f;
    lsum += p0 + p1;
    *dst = (unsigned)f2bf(p0) | ((unsigned)f2bf(p1) << 16);
}

// ---- fused masked softmax + (att @ Wh) + relu ----
// block = 16 rows; streams 32 columns/iter; un-normalized exp (exact: scores bounded,
// masked entries are exactly 0, self-loop guarantees denominator > 0).
// Double-buffered P tile in LDS, one barrier per tile, producer of tile jt+1
// overlapped with WMMA consume of tile jt. Explicit 2-stage loop, NO unrolling,
// so the hot loop stays I-cache resident.
__global__ void __launch_bounds__(256) k_attn(const float* __restrict__ adj,
                                              const unsigned short* __restrict__ WhbT,
                                              const float* __restrict__ f1,
                                              const float* __restrict__ f2,
                                              float* __restrict__ out) {
    __shared__ unsigned short s_p[2][16][32];   // double-buffered bf16 P tile
    __shared__ float s_red[16][16];
    __shared__ float s_linv[16];

    const int tid  = threadIdx.x;
    const int lane = tid & 31;
    const int wid  = tid >> 5;
    const int hi   = lane >> 4;
    const int l15  = lane & 15;
    const int row0 = blockIdx.x * 16;
    const int c0   = wid * 32;

    // producer role: thread covers (row pr, cols pc..pc+1) of the 16x32 score tile
    const int pr = tid >> 4;
    const int pc = (tid & 15) * 2;
    const float f1v = f1[row0 + pr];
    const float* adjrow = adj + (size_t)(row0 + pr) * NN + pc;

    const unsigned short* bbase0 = WhbT + (size_t)(c0 + l15) * NN + hi * 8;
    const unsigned short* bbase1 = WhbT + (size_t)(c0 + 16 + l15) * NN + hi * 8;
    const unsigned short* abase0 = &s_p[0][l15][hi * 8];
    const unsigned short* abase1 = &s_p[1][l15][hi * 8];
    unsigned int* dst0 = (unsigned int*)&s_p[0][pr][pc];
    unsigned int* dst1 = (unsigned int*)&s_p[1][pr][pc];

    v8f acc0 = {}; v8f acc1 = {};
    float lsum = 0.f;

    // prologue: produce tile 0 into buffer 0
    {
        float2 ad = *(const float2*)(adjrow);
        produce_tile(dst0, ad, f2[pc], f2[pc + 1], f1v, lsum);
    }

    // tiles 0..253 consumed, tiles 1..254 produced (two tiles per trip)
#pragma unroll 1
    for (int i = 0; i < 127; ++i) {
        const int j0 = i * 64;
        // -- stage A: consume tile (buf0), produce next tile (buf1) --
        float2 ad = *(const float2*)(adjrow + j0 + 32);
        float g0 = f2[j0 + 32 + pc], g1 = f2[j0 + 32 + pc + 1];
        __syncthreads();
        consume_tile(abase0, bbase0, bbase1, j0, acc0, acc1);
        produce_tile(dst1, ad, g0, g1, f1v, lsum);
        // -- stage B: consume tile (buf1), produce next tile (buf0) --
        ad = *(const float2*)(adjrow + j0 + 64);
        g0 = f2[j0 + 64 + pc]; g1 = f2[j0 + 64 + pc + 1];
        __syncthreads();
        consume_tile(abase1, bbase0, bbase1, j0 + 32, acc0, acc1);
        produce_tile(dst0, ad, g0, g1, f1v, lsum);
    }

    // tail: consume tile 254 (buf0), produce + consume tile 255 (buf1)
    {
        const int j0 = 254 * 32;
        float2 ad = *(const float2*)(adjrow + j0 + 32);
        float g0 = f2[j0 + 32 + pc], g1 = f2[j0 + 32 + pc + 1];
        __syncthreads();
        consume_tile(abase0, bbase0, bbase1, j0, acc0, acc1);
        produce_tile(dst1, ad, g0, g1, f1v, lsum);
        __syncthreads();
        consume_tile(abase1, bbase0, bbase1, j0 + 32, acc0, acc1);
    }

    // reduce row sums (16 partials per row)
    __syncthreads();
    s_red[pr][tid & 15] = lsum;
    __syncthreads();
    if (tid < 16) {
        float t = 0.f;
#pragma unroll
        for (int i = 0; i < 16; ++i) t += s_red[tid][i];
        s_linv[tid] = 1.0f / t;
    }
    __syncthreads();

#pragma unroll
    for (int v = 0; v < 8; ++v) {
        int m = v + hi * 8;
        float inv = s_linv[m];
        float o0 = acc0[v] * inv; o0 = (o0 > 0.f) ? o0 : 0.f;
        float o1 = acc1[v] * inv; o1 = (o1 > 0.f) ? o1 : 0.f;
        out[(size_t)(row0 + m) * DD + c0 + l15] = o0;
        out[(size_t)(row0 + m) * DD + c0 + 16 + l15] = o1;
    }
}

extern "C" void kernel_launch(void* const* d_in, const int* in_sizes, int n_in,
                              void* d_out, int out_size, void* d_ws, size_t ws_size,
                              hipStream_t stream) {
    const float* X   = (const float*)d_in[0];   // [N,D]
    const float* adj = (const float*)d_in[1];   // [N,N]
    // d_in[2] = cmt_weight (unused by reference)
    const float* W   = (const float*)d_in[3];   // [D,D]
    const float* a1  = (const float*)d_in[4];   // [D]
    const float* a2  = (const float*)d_in[5];   // [D]
    float* out = (float*)d_out;                 // [N,D]

    char* ws = (char*)d_ws;
    unsigned short* Xb   = (unsigned short*)ws; ws += (size_t)NN * DD * 2;  // 4 MB
    unsigned short* WbT  = (unsigned short*)ws; ws += (size_t)DD * DD * 2;  // 128 KB
    float*          Wh   = (float*)ws;          ws += (size_t)NN * DD * 4;  // 8 MB
    unsigned short* WhbT = (unsigned short*)ws; ws += (size_t)NN * DD * 2;  // 4 MB
    float*          f1   = (float*)ws;          ws += (size_t)NN * 4;
    float*          f2   = (float*)ws;          ws += (size_t)NN * 4;

    k_cvt_x <<<(NN * DD) / 1024, 256, 0, stream>>>(X, Xb);
    k_cvt_wT<<<DD, 256, 0, stream>>>(W, WbT);
    k_gemm1 <<<NN / 16, 256, 0, stream>>>(Xb, WbT, Wh, WhbT);
    k_f12   <<<NN / 8, 256, 0, stream>>>(Wh, a1, a2, f1, f2);
    k_attn  <<<NN / 16, 256, 0, stream>>>(adj, WhbT, f1, f2, out);
}
